// SHMAgent_69965017252757
// MI455X (gfx1250) — compile-verified, matrix operand order checked by reading, not run on hardware
//
#include <hip/hip_runtime.h>

// ---------------- problem constants ----------------
#define T_LEN 2048
#define B_SZ  32
#define D_SZ  1024
#define H_SZ  1024
#define M_SZ  24
#define ROWS  (T_LEN * B_SZ)   // 65536 tokens, row n = t*B + b
#define NCOL  160              // padded projection columns (5*24 + 1 eta + pad)
#define CLAMP_V 100.0f

// ---------------- ws layout (bytes) ----------------
#define WT_OFF  0                       // bf16 packed weights [NCOL][D]   = 327680 B
#define P_OFF   327680                  // raw projections f32 [ROWS][160] = 41943040 B
#define V4_OFF  42270720                // per-token vecs f32 [ROWS][96]   = 25165824 B
#define C_OFF   67436544                // C matrices f32 [ROWS][576]      = 150994944 B
#define Y_OFF   218431488               // y f32 [ROWS][24]                = 6291456 B

// ---------------- WMMA types ----------------
typedef __attribute__((ext_vector_type(16))) __bf16 v16bf;
typedef __attribute__((ext_vector_type(8)))  float  v8f;
typedef __attribute__((ext_vector_type(2)))  float  v2f;

union FragBF { uint4 q[2]; v16bf v; };

// ---------------- helpers ----------------
__device__ __forceinline__ int lane_id() { return threadIdx.x & 31; }

__device__ __forceinline__ float bcastf(float v, int src) {
  return __builtin_bit_cast(float,
      __builtin_amdgcn_ds_bpermute(src << 2, __builtin_bit_cast(int, v)));
}
__device__ __forceinline__ float xorf(float v, int off) {
  int l = lane_id();
  return __builtin_bit_cast(float,
      __builtin_amdgcn_ds_bpermute((l ^ off) << 2, __builtin_bit_cast(int, v)));
}
__device__ __forceinline__ float wsum32(float v) {
  v += xorf(v, 16); v += xorf(v, 8); v += xorf(v, 4);
  v += xorf(v, 2);  v += xorf(v, 1);
  return v;
}
// float -> bf16 bits, round-to-nearest-even
__device__ __forceinline__ unsigned short f2bf(float f) {
  unsigned u = __builtin_bit_cast(unsigned, f);
  unsigned r = u + 0x7FFFu + ((u >> 16) & 1u);
  return (unsigned short)(r >> 16);
}

// ============================================================
// Kernel 1: pack the 6 projection weights into bf16, N-major
// Wt[n][k], n in [0,160): 0-23 key, 24-47 query, 48-71 value,
// 72-95 vc, 96-119 shortcut, 120 eta, 121-159 zero pad.
// ============================================================
__global__ void pack_w_kernel(const float* __restrict__ kw, const float* __restrict__ qw,
                              const float* __restrict__ vw, const float* __restrict__ vcw,
                              const float* __restrict__ ew, const float* __restrict__ sw,
                              unsigned short* __restrict__ wt) {
  int idx = blockIdx.x * 256 + threadIdx.x;
  if (idx >= NCOL * D_SZ) return;
  int n = idx / D_SZ, k = idx % D_SZ;
  float val = 0.f;
  if      (n <  24) val = kw [(n      ) * D_SZ + k];
  else if (n <  48) val = qw [(n - 24 ) * D_SZ + k];
  else if (n <  72) val = vw [(n - 48 ) * D_SZ + k];
  else if (n <  96) val = vcw[(n - 72 ) * D_SZ + k];
  else if (n < 120) val = sw [(n - 96 ) * D_SZ + k];
  else if (n == 120) val = ew[k];
  wt[idx] = f2bf(val);
}

// ============================================================
// Kernel 2: LayerNorm + fused projection GEMM (bf16 WMMA)
// 64 tokens per block (256 thr = 8 waves); xn staged as bf16
// in 128 KB LDS; 4 row-tiles x 10 col-tiles, K-loop of 32.
// ============================================================
__global__ void proj_kernel(const float* __restrict__ x,
                            const float* __restrict__ g,
                            const float* __restrict__ bvec,
                            const unsigned short* __restrict__ wt,
                            float* __restrict__ P) {
  extern __shared__ unsigned char smem[];          // [64][1024] bf16
  const int lane = threadIdx.x & 31;
  const int wave = threadIdx.x >> 5;
  const int half = lane >> 4;
  const int rowbase = blockIdx.x * 64;

  // ---- stage 1: LayerNorm, write xn (bf16) to LDS ----
  for (int rr = 0; rr < 8; ++rr) {
    int row = wave * 8 + rr;
    const float* xr = x + (size_t)(rowbase + row) * D_SZ;
    float4 xv[8];
    float s = 0.f, s2 = 0.f;
#pragma unroll
    for (int i = 0; i < 8; ++i) {
      xv[i] = *(const float4*)(xr + i * 128 + lane * 4);
      s  += xv[i].x + xv[i].y + xv[i].z + xv[i].w;
      s2 += xv[i].x * xv[i].x + xv[i].y * xv[i].y
          + xv[i].z * xv[i].z + xv[i].w * xv[i].w;
    }
    s = wsum32(s); s2 = wsum32(s2);
    float mean = s * (1.f / 1024.f);
    float var  = s2 * (1.f / 1024.f) - mean * mean;
    float rstd = rsqrtf(var + 1e-5f);
    unsigned char* dst = smem + row * 2048;
#pragma unroll
    for (int i = 0; i < 8; ++i) {
      float4 gv = *(const float4*)(g    + i * 128 + lane * 4);
      float4 bv = *(const float4*)(bvec + i * 128 + lane * 4);
      float nx = (xv[i].x - mean) * rstd * gv.x + bv.x;
      float ny = (xv[i].y - mean) * rstd * gv.y + bv.y;
      float nz = (xv[i].z - mean) * rstd * gv.z + bv.z;
      float nw = (xv[i].w - mean) * rstd * gv.w + bv.w;
      uint2 pk;
      pk.x = (unsigned)f2bf(nx) | ((unsigned)f2bf(ny) << 16);
      pk.y = (unsigned)f2bf(nz) | ((unsigned)f2bf(nw) << 16);
      *(uint2*)(dst + (i * 128 + lane * 4) * 2) = pk;
    }
  }
  __syncthreads();

  // ---- stage 2: bf16 WMMA, 16x16x32 tiles ----
  const int m = lane & 15;
  for (int s = wave; s < 40; s += 8) {
    int rt = s / 10, nt = s % 10;
    v8f acc = {};
    const unsigned char*  arow = smem + (rt * 16 + m) * 2048;
    const unsigned short* bcol = wt + (size_t)(nt * 16 + m) * D_SZ;
    for (int kk = 0; kk < D_SZ; kk += 32) {
      FragBF a, b;
      // A fragment (16-bit A layout: VGPR0-3 = K half*8..+7, VGPR4-7 = K 16+half*8..+7)
      a.q[0] = *(const uint4*)(arow + (kk      + half * 8) * 2);
      a.q[1] = *(const uint4*)(arow + (kk + 16 + half * 8) * 2);
      // B fragment from N-major weights (K contiguous)
      b.q[0] = *(const uint4*)(bcol + kk      + half * 8);
      b.q[1] = *(const uint4*)(bcol + kk + 16 + half * 8);
      if (kk + 32 < D_SZ) __builtin_prefetch(bcol + kk + 32, 0, 0);
      acc = __builtin_amdgcn_wmma_f32_16x16x32_bf16(false, a.v, false, b.v,
                                                    (short)0, acc, false, false);
    }
    // D layout: lane n = lane&15, VGPR r -> M = r + half*8
    float* prow = P + (size_t)(rowbase + rt * 16 + half * 8) * NCOL + nt * 16 + m;
#pragma unroll
    for (int r = 0; r < 8; ++r) prow[(size_t)r * NCOL] = acc[r];
  }
}

// ============================================================
// Kernel 3: per-token post-processing (one wave per token):
// normalize K/Q, eta*V, shortcut+bias -> V4[96]; theta gather;
// C[i][j] = 1 + tanh(theta_i * vc_j)  (all transcendentals
// hoisted off the serial scan's critical path)
// ============================================================
__global__ void prep_kernel(const float* __restrict__ P,
                            const int* __restrict__ ri,
                            const float* __restrict__ theta_m,
                            const float* __restrict__ shb,
                            float* __restrict__ V4,
                            float* __restrict__ C) {
  const int lane = threadIdx.x & 31;
  const int wave = threadIdx.x >> 5;
  const int row  = blockIdx.x * 8 + wave;
  const bool act = lane < 24;
  const int  j   = act ? lane : 0;

  const float* pr = P + (size_t)row * NCOL;
  float kraw = pr[j],  qraw = pr[24 + j], vraw = pr[48 + j];
  float vcj  = pr[72 + j], scr = pr[96 + j], eraw = pr[120];

  float kr = act ? fmaxf(kraw, 0.f) : 0.f;
  float qr = act ? fmaxf(qraw, 0.f) : 0.f;
  float ks = wsum32(kr), qs = wsum32(qr);
  float kn = kr / (1e-5f + ks);
  float qn = qr / (1e-5f + qs);
  float eta1 = 1.f / (1.f + __expf(-eraw));
  float ev = eta1 * vraw;
  float sc = scr + (act ? shb[j] : 0.f);

  float* vr = V4 + (size_t)row * 96;
  if (act) { vr[j] = kn; vr[24 + j] = qn; vr[48 + j] = ev; vr[72 + j] = sc; }

  // theta = theta_matrix[ri[b*T + t]]  with row = t*B + b
  int b = row & (B_SZ - 1), t = row >> 5;
  int riv = ri[b * T_LEN + t];
  float th = theta_m[riv * M_SZ + j];

  float* cr = C + (size_t)row * 576;
#pragma unroll 4
  for (int i = 0; i < 24; ++i) {
    float thi = bcastf(th, i);                 // all lanes participate
    if (act) cr[i * 24 + j] = 1.f + tanhf(thi * vcj);
  }
}

// ============================================================
// Kernel 4: sequential scan — one wave per batch, lane i owns
// state row S[i][0..23] in registers. No cross-lane reduction:
// y_i = sum_j S_ij * Q_j with Q/K broadcast via ds_bpermute.
// ============================================================
__global__ void scan_kernel(const float* __restrict__ C,
                            const float* __restrict__ V4,
                            float* __restrict__ Y,
                            float* __restrict__ slast) {
  const int lane = threadIdx.x;          // blockDim = 32
  const int b    = blockIdx.x;
  const bool act = lane < 24;
  const int  i   = act ? lane : 0;

  float S[24];
#pragma unroll
  for (int jj = 0; jj < 24; ++jj) S[jj] = 0.f;

  for (int t = 0; t < T_LEN; ++t) {
    const int row = t * B_SZ + b;
    const float* cr = C + (size_t)row * 576 + i * 24;
    union { float4 q[6]; float f[24]; } cu;
#pragma unroll
    for (int u = 0; u < 6; ++u) cu.q[u] = *(const float4*)(cr + u * 4);

    const float* vr = V4 + (size_t)row * 96;
    float knl = vr[i], ql = vr[24 + i];
    float ev  = vr[48 + i], sc = vr[72 + i];

    if (t + 1 < T_LEN) {   // prefetch next step off the critical path
      __builtin_prefetch(C + (size_t)(row + B_SZ) * 576 + i * 24, 0, 0);
      __builtin_prefetch(V4 + (size_t)(row + B_SZ) * 96, 0, 0);
    }

    float y = 0.f;
#pragma unroll
    for (int jj = 0; jj < 24; ++jj) {
      float kj = bcastf(knl, jj);
      float qj = bcastf(ql,  jj);
      float sv = S[jj] * cu.f[jj] + ev * kj;
      sv = fminf(fmaxf(sv, -CLAMP_V), CLAMP_V);
      S[jj] = sv;
      y += sv * qj;
    }
    if (act) Y[(size_t)row * 24 + i] = y + sc;
  }
  if (act) {
#pragma unroll
    for (int jj = 0; jj < 24; ++jj)
      slast[(size_t)b * 576 + i * 24 + jj] = S[jj];
  }
}

// ============================================================
// Kernel 5: z = y @ out_w^T + out_b via exact f32 WMMA
// (K = 24 = 6 x 16x16x4 steps). One 16x16 tile per wave.
// ============================================================
__global__ void out_kernel(const float* __restrict__ Y,
                           const float* __restrict__ ow,
                           const float* __restrict__ ob,
                           float* __restrict__ Z) {
  const int lane = threadIdx.x & 31;
  const int wave = threadIdx.x >> 5;
  const int half = lane >> 4;
  const int m    = lane & 15;
  const int tile = blockIdx.x * 8 + wave;     // ROWS/16 * H/16 = 262144 tiles
  const int rt = tile >> 6;                   // token tile
  const int ht = tile & 63;                   // H tile

  v8f acc = {};
  const float* ya = Y  + (size_t)(rt * 16 + m) * 24 + 2 * half;
  const float* ba = ow + (size_t)(ht * 16 + m) * 24 + 2 * half;
#pragma unroll
  for (int kc = 0; kc < 24; kc += 4) {
    // f32 A 16x4 layout: VGPR0 K=kc+2*half, VGPR1 K=kc+2*half+1
    v2f a = __builtin_bit_cast(v2f, *(const float2*)(ya + kc));
    v2f b = __builtin_bit_cast(v2f, *(const float2*)(ba + kc));
    acc = __builtin_amdgcn_wmma_f32_16x16x4_f32(false, a, false, b,
                                                (short)0, acc, false, false);
  }
  float bias = ob[ht * 16 + m];
  float* zp = Z + (size_t)(rt * 16 + half * 8) * H_SZ + ht * 16 + m;
#pragma unroll
  for (int r = 0; r < 8; ++r) zp[(size_t)r * H_SZ] = acc[r] + bias;
}

// ============================================================
extern "C" void kernel_launch(void* const* d_in, const int* in_sizes, int n_in,
                              void* d_out, int out_size, void* d_ws, size_t ws_size,
                              hipStream_t stream) {
  (void)in_sizes; (void)n_in; (void)out_size; (void)ws_size;
  const float* x   = (const float*)d_in[0];
  const int*   ri  = (const int*)  d_in[1];
  const float* kw  = (const float*)d_in[2];
  const float* qw  = (const float*)d_in[3];
  const float* vw  = (const float*)d_in[4];
  const float* vcw = (const float*)d_in[5];
  const float* ew  = (const float*)d_in[6];
  const float* sw  = (const float*)d_in[7];
  const float* shb = (const float*)d_in[8];
  const float* ow  = (const float*)d_in[9];
  const float* ob  = (const float*)d_in[10];
  const float* thm = (const float*)d_in[11];
  const float* lng = (const float*)d_in[12];
  const float* lnb = (const float*)d_in[13];

  float* z = (float*)d_out;
  unsigned char* ws = (unsigned char*)d_ws;
  unsigned short* wt = (unsigned short*)(ws + WT_OFF);
  float* P  = (float*)(ws + P_OFF);
  float* V4 = (float*)(ws + V4_OFF);
  float* C  = (float*)(ws + C_OFF);
  float* Y  = (float*)(ws + Y_OFF);

  pack_w_kernel<<<(NCOL * D_SZ) / 256, 256, 0, stream>>>(kw, qw, vw, vcw, ew, sw, wt);
  proj_kernel  <<<ROWS / 64, 256, 64 * D_SZ * 2, stream>>>(x, lng, lnb, wt, P);
  prep_kernel  <<<ROWS / 8, 256, 0, stream>>>(P, ri, thm, shb, V4, C);
  scan_kernel  <<<B_SZ, 32, 0, stream>>>(C, V4, Y, z + (size_t)ROWS * H_SZ);
  out_kernel   <<<(ROWS / 16) * (H_SZ / 16) / 8, 256, 0, stream>>>(Y, ow, ob, z);
}
